// GNNNet_72086731096485
// MI455X (gfx1250) — compile-verified, matrix operand order checked by reading, not run on hardware
//
#include <hip/hip_runtime.h>
#include <hip/hip_bf16.h>
#include <math.h>

#define IN_DIM 128
#define HID    64
#define OUT_DIM 64
#define GAT_SLOPE 0.2f
#define ACT_SLOPE 0.01f

typedef __attribute__((ext_vector_type(16))) __bf16 v16bf;
typedef __attribute__((ext_vector_type(8)))  __bf16 v8bf;
typedef __attribute__((ext_vector_type(8)))  float  v8f;

__device__ __forceinline__ float lrelu(float v, float s) { return v > 0.f ? v : v * s; }

// monotone mapping float -> uint32 so unsigned atomicMax == float max
__device__ __forceinline__ unsigned f2mono(float f) {
    unsigned u = __float_as_uint(f);
    return (u & 0x80000000u) ? ~u : (u | 0x80000000u);
}
__device__ __forceinline__ float mono2f(unsigned m) {
    unsigned u = (m & 0x80000000u) ? (m ^ 0x80000000u) : ~m;
    return __uint_as_float(u);
}

// pack W[k,n] (fp32, [K,64]) -> Wt[n,k] (bf16, [64,K])
__global__ void pack_wt(const float* __restrict__ W, __bf16* __restrict__ Wt, int K) {
    int idx = blockIdx.x * blockDim.x + threadIdx.x;
    if (idx >= 64 * K) return;
    int n = idx / K, k = idx - n * K;
    Wt[idx] = (__bf16)W[(size_t)k * 64 + n];
}

// ---------------------------------------------------------------------------
// WMMA GEMM: out[m, 0..63] = act( sum_k A[row(m), k] * W[k, n] + bias[n] )
// A fp32 [M,K]; Wt bf16 [64,K] (pre-transposed), staged in LDS (row stride
// K+8 bf16 to avoid bank conflicts). Block = 256 threads (8 waves); each
// block does a 32(M) x 64(N) tile; each wave one 16x16 WMMA tile; K-loop
// fully unrolled (bf16 WMMA, K-step 32). rowidx != nullptr => gather rows.
// All loads unconditional (row clamped); only stores are guarded, so EXEC
// stays full around the WMMAs.
// ---------------------------------------------------------------------------
template <int K>
__global__ void wmma_gemm64(const float* __restrict__ A,
                            const __bf16* __restrict__ Wt,
                            const float* __restrict__ bias,
                            const int* __restrict__ rowidx,
                            float* __restrict__ out,
                            int M, float slope, int do_act) {
    constexpr int LDK = K + 8;               // padded LDS row stride (bf16 elems)
    extern __shared__ char smem[];
    __bf16* sW = (__bf16*)smem;              // [64][LDK]

    // cooperative stage of Wt into LDS
    for (int idx = threadIdx.x; idx < 64 * K; idx += 256) {
        int n = idx / K, k = idx - n * K;
        sW[n * LDK + k] = Wt[idx];
    }
    __syncthreads();

    const int lane = threadIdx.x & 31;
    const int wave = threadIdx.x >> 5;
    const int half = lane >> 4;              // 0: lanes 0-15, 1: lanes 16-31
    const int lm   = lane & 15;

    const int m0 = blockIdx.x * 32 + (wave >> 2) * 16;  // row tile base
    const int n0 = (wave & 3) * 16;                     // col tile base

    const int row = m0 + lm;                 // logical output row for A fetch
    int rc = row < M ? row : (M - 1);        // clamp: loads always valid
    const int arow = rowidx ? rowidx[rc] : rc;

    const float*  __restrict__ arow_p = A + (size_t)arow * (size_t)K;
    const __bf16* __restrict__ swc    = sW + (size_t)(n0 + lm) * LDK;

    v8f acc = {};

    #pragma unroll
    for (int kc = 0; kc < K; kc += 32) {
        v16bf a, b;
        // A: 16-bit A-matrix 16x32 layout; element j: k = kc+(j/8)*16+half*8+(j%8)
        #pragma unroll
        for (int g = 0; g < 2; ++g) {
            const int kb = kc + g * 16 + half * 8;
            float4 f0 = *(const float4*)(arow_p + kb);
            float4 f1 = *(const float4*)(arow_p + kb + 4);
            const int j0 = g * 8;
            a[j0 + 0] = (__bf16)f0.x; a[j0 + 1] = (__bf16)f0.y;
            a[j0 + 2] = (__bf16)f0.z; a[j0 + 3] = (__bf16)f0.w;
            a[j0 + 4] = (__bf16)f1.x; a[j0 + 5] = (__bf16)f1.y;
            a[j0 + 6] = (__bf16)f1.z; a[j0 + 7] = (__bf16)f1.w;
        }
        // B: column n = n0+lm, same K mapping; 16B ds_load_b128 per group
        #pragma unroll
        for (int g = 0; g < 2; ++g) {
            const int kb = kc + g * 16 + half * 8;
            v8bf bw = *(const v8bf*)(swc + kb);
            const int j0 = g * 8;
            #pragma unroll
            for (int t = 0; t < 8; ++t) b[j0 + t] = bw[t];
        }
        acc = __builtin_amdgcn_wmma_f32_16x16x32_bf16(
            /*neg_a=*/false, a, /*neg_b=*/false, b,
            /*c_mod=*/(short)0, acc, /*reuse_a=*/false, /*reuse_b=*/false);
    }

    // C/D layout: VGPR r -> m = m0 + r + 8*half, n = n0 + lm
    #pragma unroll
    for (int r = 0; r < 8; ++r) {
        const int m = m0 + r + half * 8;
        if (m < M) {
            const int n = n0 + lm;
            float v = acc[r];
            if (bias) v += bias[n];
            if (do_act) v = lrelu(v, slope);
            out[(size_t)m * 64 + n] = v;
        }
    }
}

// per-node attention logits: alpha_s[i] = hW[i,:].a_src ; alpha_d[i] = hW[i,:].a_dst
__global__ void node_alpha(const float* __restrict__ hW,
                           const float* __restrict__ a_src,
                           const float* __restrict__ a_dst,
                           float* __restrict__ as_, float* __restrict__ ad_, int N) {
    int i = blockIdx.x * blockDim.x + threadIdx.x;
    if (i >= N) return;
    const float4* row = (const float4*)(hW + (size_t)i * HID);
    const float4* vs  = (const float4*)a_src;
    const float4* vd  = (const float4*)a_dst;
    float s = 0.f, d = 0.f;
    #pragma unroll
    for (int q = 0; q < HID / 4; ++q) {
        float4 h = row[q], ws = vs[q], wd = vd[q];
        s += h.x * ws.x + h.y * ws.y + h.z * ws.z + h.w * ws.w;
        d += h.x * wd.x + h.y * wd.y + h.z * wd.z + h.w * wd.w;
    }
    as_[i] = s; ad_[i] = d;
}

// self-loop logit + init segment-max with it
__global__ void init_emax(const float* __restrict__ as_, const float* __restrict__ ad_,
                          float* __restrict__ eself, unsigned* __restrict__ emax, int N) {
    int i = blockIdx.x * blockDim.x + threadIdx.x;
    if (i >= N) return;
    float es = lrelu(as_[i] + ad_[i], GAT_SLOPE);
    eself[i] = es;
    emax[i]  = f2mono(es);
}

__global__ void edge_emax(const float* __restrict__ as_, const float* __restrict__ ad_,
                          const int* __restrict__ src, const int* __restrict__ dst,
                          unsigned* __restrict__ emax, int E) {
    int e = blockIdx.x * blockDim.x + threadIdx.x;
    if (e >= E) return;
    float ev = lrelu(as_[src[e]] + ad_[dst[e]], GAT_SLOPE);
    atomicMax(&emax[dst[e]], f2mono(ev));
}

__global__ void init_denom(const float* __restrict__ eself, const unsigned* __restrict__ emax,
                           float* __restrict__ denom, int N) {
    int i = blockIdx.x * blockDim.x + threadIdx.x;
    if (i >= N) return;
    denom[i] = expf(eself[i] - mono2f(emax[i]));
}

__global__ void edge_ex(const float* __restrict__ as_, const float* __restrict__ ad_,
                        const int* __restrict__ src, const int* __restrict__ dst,
                        const unsigned* __restrict__ emax,
                        float* __restrict__ exbuf, float* __restrict__ denom, int E) {
    int e = blockIdx.x * blockDim.x + threadIdx.x;
    if (e >= E) return;
    int d = dst[e];
    float ev = lrelu(as_[src[e]] + ad_[d], GAT_SLOPE);
    float ex = expf(ev - mono2f(emax[d]));
    exbuf[e] = ex;
    atomicAdd(&denom[d], ex);
}

// eself -> self-loop attention weight; exbuf -> edge attention weight
__global__ void self_alpha(float* __restrict__ eself, const unsigned* __restrict__ emax,
                           const float* __restrict__ denom, int N) {
    int i = blockIdx.x * blockDim.x + threadIdx.x;
    if (i >= N) return;
    eself[i] = expf(eself[i] - mono2f(emax[i])) / denom[i];
}

__global__ void edge_alpha(float* __restrict__ exbuf, const float* __restrict__ denom,
                           const int* __restrict__ dst, int E) {
    int e = blockIdx.x * blockDim.x + threadIdx.x;
    if (e >= E) return;
    exbuf[e] /= denom[dst[e]];
}

// agg[i,:] = hW[i,:] * alpha_self[i]
__global__ void init_agg(const float* __restrict__ hW, const float* __restrict__ aself,
                         float* __restrict__ agg, int total) {
    int idx = blockIdx.x * blockDim.x + threadIdx.x;
    if (idx >= total) return;
    int i = idx >> 6;
    agg[idx] = hW[idx] * aself[i];
}

// agg[dst,:] += hW[src,:] * alpha_e  (one thread per (edge, col))
__global__ void edge_agg(const float* __restrict__ hW, const float* __restrict__ alpha_e,
                         const int* __restrict__ src, const int* __restrict__ dst,
                         float* __restrict__ agg, int total) {
    int idx = blockIdx.x * blockDim.x + threadIdx.x;
    if (idx >= total) return;
    int e = idx >> 6, c = idx & 63;
    atomicAdd(&agg[(size_t)dst[e] * 64 + c], hW[(size_t)src[e] * 64 + c] * alpha_e[e]);
}

// h_next = leaky_relu(agg + b, 0.01)
__global__ void conv_finalize(const float* __restrict__ agg, const float* __restrict__ b,
                              float* __restrict__ h_next, int total) {
    int idx = blockIdx.x * blockDim.x + threadIdx.x;
    if (idx >= total) return;
    h_next[idx] = lrelu(agg[idx] + b[idx & 63], ACT_SLOPE);
}

// first_idx[g] = lower_bound(batch, g)
__global__ void first_index(const int* __restrict__ batch, int N, int G,
                            int* __restrict__ fidx) {
    int g = blockIdx.x * blockDim.x + threadIdx.x;
    if (g >= G) return;
    int lo = 0, hi = N;
    while (lo < hi) { int mid = (lo + hi) >> 1; if (batch[mid] < g) lo = mid + 1; else hi = mid; }
    fidx[g] = lo;
}

__global__ void zero_f32(float* __restrict__ p, int n) {
    int i = blockIdx.x * blockDim.x + threadIdx.x;
    if (i < n) p[i] = 0.f;
}

__global__ void graph_count(const int* __restrict__ batch, float* __restrict__ cnt, int N) {
    int i = blockIdx.x * blockDim.x + threadIdx.x;
    if (i < N) atomicAdd(&cnt[batch[i]], 1.0f);
}

extern "C" void kernel_launch(void* const* d_in, const int* in_sizes, int n_in,
                              void* d_out, int out_size, void* d_ws, size_t ws_size,
                              hipStream_t stream) {
    const float* x      = (const float*)d_in[0];
    const int*   eidx   = (const int*)d_in[1];
    const int*   batch  = (const int*)d_in[2];
    // d_in[3] = num_graphs (device scalar; G derived from out_size instead)
    const float* W_emb  = (const float*)d_in[4];
    const float* b_emb  = (const float*)d_in[5];
    const float* W_c[2]    = {(const float*)d_in[6],  (const float*)d_in[10]};
    const float* a_src[2]  = {(const float*)d_in[7],  (const float*)d_in[11]};
    const float* a_dst[2]  = {(const float*)d_in[8],  (const float*)d_in[12]};
    const float* b_c[2]    = {(const float*)d_in[9],  (const float*)d_in[13]};
    const float* W_head = (const float*)d_in[14];
    const float* b_head = (const float*)d_in[15];

    const int N = in_sizes[0] / IN_DIM;
    const int E = in_sizes[1] / 2;
    const int G = out_size / (OUT_DIM + 1);   // out = [G,64] ++ [G,1]
    const int* src = eidx;
    const int* dst = eidx + E;

    // workspace carve-up
    char* ws = (char*)d_ws;
    const size_t nodeMat = (size_t)N * HID * sizeof(float);
    float*    bufA   = (float*)ws;    ws += nodeMat;   // current node features
    float*    bufB   = (float*)ws;    ws += nodeMat;   // hW
    float*    bufC   = (float*)ws;    ws += nodeMat;   // aggregation
    float*    as_    = (float*)ws;    ws += (size_t)N * sizeof(float);
    float*    ad_    = (float*)ws;    ws += (size_t)N * sizeof(float);
    float*    eself  = (float*)ws;    ws += (size_t)N * sizeof(float);
    unsigned* emax   = (unsigned*)ws; ws += (size_t)N * sizeof(unsigned);
    float*    denom  = (float*)ws;    ws += (size_t)N * sizeof(float);
    float*    exbuf  = (float*)ws;    ws += (size_t)E * sizeof(float);
    int*      fidx   = (int*)ws;      ws += (size_t)G * sizeof(int);
    __bf16*   WtEmb  = (__bf16*)ws;   ws += (size_t)64 * IN_DIM * sizeof(__bf16);
    __bf16*   WtC[2];
    WtC[0] = (__bf16*)ws;             ws += (size_t)64 * HID * sizeof(__bf16);
    WtC[1] = (__bf16*)ws;             ws += (size_t)64 * HID * sizeof(__bf16);
    __bf16*   WtHead = (__bf16*)ws;   ws += (size_t)64 * HID * sizeof(__bf16);

    const dim3 blk(256);
    const int gN    = (N + 255) / 256;
    const int gE    = (E + 255) / 256;
    const int gG    = (G + 255) / 256;
    const int gN32  = (N + 31) / 32;
    const int nc    = N * HID;
    const int gNC   = (nc + 255) / 256;
    const int ec    = E * HID;
    const int gEC   = (ec + 255) / 256;
    const size_t sh128 = (size_t)64 * (IN_DIM + 8) * sizeof(__bf16);
    const size_t sh64  = (size_t)64 * (HID + 8)    * sizeof(__bf16);

    // pre-pack all weight matrices to transposed bf16
    pack_wt<<<(64 * IN_DIM + 255) / 256, blk, 0, stream>>>(W_emb,  WtEmb,  IN_DIM);
    pack_wt<<<(64 * HID   + 255) / 256, blk, 0, stream>>>(W_c[0], WtC[0], HID);
    pack_wt<<<(64 * HID   + 255) / 256, blk, 0, stream>>>(W_c[1], WtC[1], HID);
    pack_wt<<<(64 * HID   + 255) / 256, blk, 0, stream>>>(W_head, WtHead, HID);

    // embedding MLP: bufA = leaky_relu(x @ W_emb + b_emb, 0.01)
    wmma_gemm64<IN_DIM><<<gN32, blk, sh128, stream>>>(x, WtEmb, b_emb, nullptr,
                                                      bufA, N, ACT_SLOPE, 1);

    for (int layer = 0; layer < 2; ++layer) {
        // hW = h @ W   (bias applied after aggregation, per reference)
        wmma_gemm64<HID><<<gN32, blk, sh64, stream>>>(bufA, WtC[layer], nullptr, nullptr,
                                                      bufB, N, 0.f, 0);
        node_alpha<<<gN, blk, 0, stream>>>(bufB, a_src[layer], a_dst[layer], as_, ad_, N);
        init_emax<<<gN, blk, 0, stream>>>(as_, ad_, eself, emax, N);
        edge_emax<<<gE, blk, 0, stream>>>(as_, ad_, src, dst, emax, E);
        init_denom<<<gN, blk, 0, stream>>>(eself, emax, denom, N);
        edge_ex<<<gE, blk, 0, stream>>>(as_, ad_, src, dst, emax, exbuf, denom, E);
        self_alpha<<<gN, blk, 0, stream>>>(eself, emax, denom, N);
        edge_alpha<<<gE, blk, 0, stream>>>(exbuf, denom, dst, E);
        init_agg<<<gNC, blk, 0, stream>>>(bufB, eself, bufC, nc);
        edge_agg<<<gEC, blk, 0, stream>>>(bufB, exbuf, src, dst, bufC, ec);
        conv_finalize<<<gNC, blk, 0, stream>>>(bufC, b_c[layer], bufA, nc);
    }

    // readout: first node per graph, then head GEMM into d_out[0 .. G*64)
    first_index<<<gG, blk, 0, stream>>>(batch, N, G, fidx);
    wmma_gemm64<HID><<<(G + 31) / 32, blk, sh64, stream>>>(bufA, WtHead, b_head, fidx,
                                                           (float*)d_out, G, 0.f, 0);

    // num_nodes per graph -> d_out[G*64 .. G*64 + G)
    float* cnt = (float*)d_out + (size_t)G * OUT_DIM;
    zero_f32<<<gG, blk, 0, stream>>>(cnt, G);
    graph_count<<<gN, blk, 0, stream>>>(batch, cnt, N);
}
